// MADF_56573309222923
// MI455X (gfx1250) — matrix operand order repaired and checked by reading, not compiled, not run on hardware
//
#include <hip/hip_runtime.h>

typedef __attribute__((ext_vector_type(16))) __bf16 v16bf;
typedef __attribute__((ext_vector_type(8)))  __bf16 v8bf;
typedef __attribute__((ext_vector_type(8)))  float  v8f;

#define B_    2
#define CMIN  64
#define CMOUT 64
#define CINE  32
#define COUTE 32
#define CIJ   288            // CINE*3*3
#define F_    9216           // COUTE*CIJ
#define NPIX  4096           // 64*64

union U16 { v16bf v; v8bf h[2]; };

// ---------------- Phase 0: w_f fp32 -> bf16 ----------------
__global__ __launch_bounds__(256) void k0_cvt_wf(const float* __restrict__ w_f,
                                                 __bf16* __restrict__ wf_bf) {
  int q = blockIdx.x * 256 + threadIdx.x;
  if (q < F_ * CMOUT) wf_bf[q] = (__bf16)w_f[q];
}

// ---------------- Phase 1: 3x3 conv + bias + relu ----------------
// writes m_l (fp32, NCHW) to d_out[0..524287] and bf16 copy pixel-major [b][p][c]
__global__ __launch_bounds__(256) void k1_conv(const float* __restrict__ x,
                                               const float* __restrict__ w_m,
                                               const float* __restrict__ b_m,
                                               float* __restrict__ m_out,
                                               __bf16* __restrict__ m_bf) {
  int q  = blockIdx.x * 256 + threadIdx.x;   // < B_*CMOUT*NPIX = 524288
  int b  = q >> 18;
  int co = (q >> 12) & 63;
  int p  = q & 4095;
  int hh = p >> 6, ww = p & 63;
  float acc = b_m[co];
  const float* wb = w_m + co * CMIN * 9;
  const float* xb = x + (size_t)b * CMIN * NPIX;
  for (int ci = 0; ci < CMIN; ++ci) {
    const float* xc = xb + ci * NPIX;
    const float* wc = wb + ci * 9;
#pragma unroll
    for (int dy = 0; dy < 3; ++dy) {
      int yy = hh + dy - 1;
      if ((unsigned)yy >= 64u) continue;
#pragma unroll
      for (int dx = 0; dx < 3; ++dx) {
        int xx = ww + dx - 1;
        if ((unsigned)xx >= 64u) continue;
        acc = fmaf(xc[yy * 64 + xx], wc[dy * 3 + dx], acc);
      }
    }
  }
  float r = fmaxf(acc, 0.f);
  m_out[q] = r;
  m_bf[(size_t)(b * NPIX + p) * CMOUT + co] = (__bf16)r;
}

// ---------------- Phase 2: fused filter-gen GEMM + dynamic conv ----------------
// block = 64 threads (2 waves); each block: half a row (32 pixels) of one (b,h).
// Per wave: 16 pixels. e_raw[b][o][p] = relu( sum_cij (w_f@m + b_f) * patch )
__global__ __launch_bounds__(64) void k2_dynconv(const float* __restrict__ e_in,
                                                 const float* __restrict__ b_f,
                                                 const __bf16* __restrict__ wf_bf,
                                                 const __bf16* __restrict__ m_bf,
                                                 float* __restrict__ e_raw) {
  __shared__ float  Pp[32][292];   // patches, padded stride (292%64 -> conflict-free)
  __shared__ __bf16 Mt[32][72];    // m tile, padded stride

  const int tid = threadIdx.x;
  const int gid = blockIdx.x;      // 0..255
  const int b   = gid >> 7;
  const int rem = gid & 127;
  const int h   = rem >> 1;
  const int wbase = (rem & 1) * 32;

  // build patches: cij = c*9 + i*3 + j  (c-major, matches reference reshape)
  for (int e = tid; e < 32 * CIJ; e += 64) {
    int pl  = e / CIJ;
    int cij = e - pl * CIJ;
    int c = cij / 9;
    int r = cij - c * 9;
    int i = r / 3, j = r - i * 3;
    int hh = h + i - 1;
    int ww = wbase + pl + j - 1;
    float v = 0.f;
    if ((unsigned)hh < 64u && (unsigned)ww < 64u)
      v = e_in[((b * CINE + c) * 64 + hh) * 64 + ww];
    Pp[pl][cij] = v;
  }
  // copy m tile: 32 pixels x 32 ch (bf16) as dwords
  for (int e = tid; e < 32 * 16; e += 64) {
    int pl = e >> 4, ch = e & 15;
    const unsigned* src =
        (const unsigned*)(m_bf + (size_t)((b * NPIX + h * 64 + wbase + pl) * CMOUT));
    ((unsigned*)&Mt[pl][0])[ch] = src[ch];
  }
  __syncthreads();

  const int lane = tid & 31;
  const int wv   = tid >> 5;
  const int hi   = lane >> 4;          // which half-wave
  const int pl   = wv * 16 + (lane & 15);

  // B matrix regs (K x N, bf16): lane holds column n=lane%16, K = hi*16 + 0..15
  U16 Bm0, Bm1;
  const __bf16* mrow = &Mt[pl][0];
  Bm0.h[0] = *(const v8bf*)(mrow + hi * 16);
  Bm0.h[1] = *(const v8bf*)(mrow + hi * 16 + 8);
  Bm1.h[0] = *(const v8bf*)(mrow + 32 + hi * 16);
  Bm1.h[1] = *(const v8bf*)(mrow + 32 + hi * 16 + 8);

  const float* prowBase = &Pp[pl][0];

  for (int o = 0; o < COUTE; ++o) {
    float acc = 0.f;
    const int fbase = o * CIJ;
    for (int blk = 0; blk < 18; ++blk) {
      // A matrix (M x K): lane holds row m=lane%16; K = hi*16 + 0..15 (+32)
      const __bf16* arow = wf_bf + (size_t)(fbase + blk * 16 + (lane & 15)) * CMOUT;
      U16 A0, A1;
      A0.h[0] = *(const v8bf*)(arow + hi * 16);
      A0.h[1] = *(const v8bf*)(arow + hi * 16 + 8);
      A1.h[0] = *(const v8bf*)(arow + 32 + hi * 16);
      A1.h[1] = *(const v8bf*)(arow + 32 + hi * 16 + 8);

      v8f c = {0.f, 0.f, 0.f, 0.f, 0.f, 0.f, 0.f, 0.f};
      c = __builtin_amdgcn_wmma_f32_16x16x32_bf16(false, A0.v, false, Bm0.v,
                                                  (short)0, c, false, false);
      c = __builtin_amdgcn_wmma_f32_16x16x32_bf16(false, A1.v, false, Bm1.v,
                                                  (short)0, c, false, false);

      // epilogue: D row m = r + 8*hi -> cij = blk*16 + 8*hi + r (8 consecutive)
      const float4* bp = (const float4*)(b_f + fbase + blk * 16 + hi * 8);
      float4 bv0 = bp[0], bv1 = bp[1];
      const float4* pp = (const float4*)(prowBase + blk * 16 + hi * 8);
      float4 p0 = pp[0], p1 = pp[1];
      acc += (c[0] + bv0.x) * p0.x + (c[1] + bv0.y) * p0.y +
             (c[2] + bv0.z) * p0.z + (c[3] + bv0.w) * p0.w +
             (c[4] + bv1.x) * p1.x + (c[5] + bv1.y) * p1.y +
             (c[6] + bv1.z) * p1.z + (c[7] + bv1.w) * p1.w;
    }
    // fold the two half-wave partials (SWAPX16 swizzle), relu, store
    float other = __int_as_float(
        __builtin_amdgcn_ds_swizzle(__float_as_int(acc), 0x401F));
    float tot = acc + other;
    if (hi == 0) {
      e_raw[(size_t)(b * COUTE + o) * NPIX + h * 64 + wbase + wv * 16 + lane] =
          fmaxf(tot, 0.f);
    }
  }
}

// ---------------- Phase 3: per-channel mean / inv-std ----------------
__global__ __launch_bounds__(256) void k3_stats(const float* __restrict__ e_raw,
                                                float* __restrict__ stats) {
  __shared__ float ss[256], ss2[256];
  int o = blockIdx.x, tid = threadIdx.x;
  float s = 0.f, s2 = 0.f;
  for (int i = tid; i < B_ * NPIX; i += 256) {
    int b = i >> 12, p = i & 4095;
    float v = e_raw[(size_t)(b * COUTE + o) * NPIX + p];
    s += v; s2 += v * v;
  }
  ss[tid] = s; ss2[tid] = s2;
  __syncthreads();
  for (int st = 128; st > 0; st >>= 1) {
    if (tid < st) { ss[tid] += ss[tid + st]; ss2[tid] += ss2[tid + st]; }
    __syncthreads();
  }
  if (tid == 0) {
    float n = (float)(B_ * NPIX);
    float mean = ss[0] / n;
    float var = ss2[0] / n - mean * mean;
    stats[o * 2 + 0] = mean;
    stats[o * 2 + 1] = rsqrtf(var + 1e-5f);
  }
}

// ---------------- Phase 4: normalize ----------------
__global__ __launch_bounds__(256) void k4_norm(const float* __restrict__ e_raw,
                                               const float* __restrict__ stats,
                                               float* __restrict__ out) {
  int idx = blockIdx.x * 256 + threadIdx.x;   // < 262144, layout == NCHW flat
  int o = (idx >> 12) & 31;
  out[idx] = (e_raw[idx] - stats[o * 2]) * stats[o * 2 + 1];
}

extern "C" void kernel_launch(void* const* d_in, const int* in_sizes, int n_in,
                              void* d_out, int out_size, void* d_ws, size_t ws_size,
                              hipStream_t stream) {
  const float* m_l_1 = (const float*)d_in[0];
  const float* e_l_1 = (const float*)d_in[1];
  const float* w_m   = (const float*)d_in[2];
  const float* b_m   = (const float*)d_in[3];
  const float* w_f   = (const float*)d_in[4];
  const float* b_f   = (const float*)d_in[5];
  float* out = (float*)d_out;

  char* ws = (char*)d_ws;
  __bf16* m_bf  = (__bf16*)(ws);                                   // 1,048,576 B
  __bf16* wf_bf = (__bf16*)(ws + 1048576);                         // 1,179,648 B
  float*  e_raw = (float*)(ws + 1048576 + 1179648);                // 1,048,576 B
  float*  stats = (float*)(ws + 1048576 + 1179648 + 1048576);      //       256 B

  // Phase 0: w_f -> bf16
  k0_cvt_wf<<<(F_ * CMOUT + 255) / 256, 256, 0, stream>>>(w_f, wf_bf);
  // Phase 1: conv3x3+relu -> out[:524288] and bf16 pixel-major scratch
  k1_conv<<<(B_ * CMOUT * NPIX) / 256, 256, 0, stream>>>(m_l_1, w_m, b_m, out, m_bf);
  // Phase 2: fused WMMA filter-gen + dynamic conv + relu -> e_raw
  k2_dynconv<<<B_ * 64 * 2, 64, 0, stream>>>(e_l_1, b_f, wf_bf, m_bf, e_raw);
  // Phase 3: BN stats
  k3_stats<<<COUTE, 256, 0, stream>>>(e_raw, stats);
  // Phase 4: normalize -> out[524288:]
  k4_norm<<<(B_ * COUTE * NPIX) / 256, 256, 0, stream>>>(e_raw, stats, out + 524288);
}
